// Splitter2_32083405701824
// MI455X (gfx1250) — matrix operand order repaired and checked by reading, not compiled, not run on hardware
//
#include <hip/hip_runtime.h>

typedef __attribute__((ext_vector_type(16))) _Float16 v16h;
typedef __attribute__((ext_vector_type(8)))  float    v8f;

#define HH 512
#define WW 512
#define PLANE (HH * WW)
#define TW 64          // output tile width per block
#define TH 32          // output tile height per block
#define LR 36          // LDS rows  (TH + 4 halo)
#define LC 68          // LDS cols  (TW + 4 halo)
#define LSTRIDE 72     // padded LDS row stride (f16 elements)

__global__ __launch_bounds__(256)
void hsl_dualconv_wmma(const float* __restrict__ x,
                       const float* __restrict__ w1, const float* __restrict__ b1,
                       const float* __restrict__ w2, const float* __restrict__ b2,
                       float* __restrict__ out_xl, float* __restrict__ out_xr)
{
    __shared__ _Float16 lds[3][LR][LSTRIDE];

    const int tid  = threadIdx.x;
    const int tX0  = blockIdx.x * TW;
    const int tY0  = blockIdx.y * TH;
    const int bimg = blockIdx.z;
    const float* xb = x + (size_t)bimg * 3 * PLANE;

    // ---- Phase 1: global f32 RGB -> HSL (VALU) -> f16 tile in LDS (with zero halo) ----
    for (int ti = tid; ti < LR * LC; ti += 256) {
        const int row = ti / LC, col = ti % LC;
        const int gy = tY0 - 2 + row, gx = tX0 - 2 + col;
        float hh = 0.f, ss = 0.f, ll = 0.f;
        if (gy >= 0 && gy < HH && gx >= 0 && gx < WW) {
            const int p = gy * WW + gx;
            const float r  = xb[p];
            const float g  = xb[PLANE + p];
            const float bl = xb[2 * PLANE + p];
            const float cmax = fmaxf(r, fmaxf(g, bl));
            const float cmin = fminf(r, fminf(g, bl));
            const float delta = cmax - cmin;
            const float inv_d = 1.0f / ((delta == 0.f) ? 1.f : delta);  // one divide
            int idx;
            if      (delta == 0.f) idx = 3;
            else if (r == cmax)    idx = 0;   // argmax: first occurrence
            else if (g == cmax)    idx = 1;
            else                   idx = 2;
            // (g-b)/delta in [-1,1]  =>  mod(v,6) == (v<0 ? v+6 : v); no fmod needed
            float h0 = (g - bl) * inv_d; if (h0 < 0.f) h0 += 6.f;
            const float h1 = (bl - r) * inv_d + 2.f;
            const float h2 = (r - g)  * inv_d + 4.f;
            const float hv = (idx == 0) ? h0 : (idx == 1) ? h1 : (idx == 2) ? h2 : 0.f;
            hh = hv * (1.f / 6.f);
            ll = 0.5f * (cmax + cmin);
            const float denom = (ll <= 0.5f) ? (2.f * ll) : (2.f - 2.f * ll);
            const float safe_den = (denom == 0.f) ? 1.f : denom;
            ss = (ll > 0.f && ll < 1.f) ? (delta / safe_den) : 0.f;
        }
        lds[0][row][col] = (_Float16)hh;
        lds[1][row][col] = (_Float16)ss;
        lds[2][row][col] = (_Float16)ll;
    }
    __syncthreads();

    // ---- Per-wave setup (wave32) ----
    const int lane = tid & 31;
    const int wave = tid >> 5;          // 0..7
    const int wx   = wave & 3;          // x sub-tile (16 px each)
    const int wy   = wave >> 2;         // y sub-tile (16 rows each)
    const int n    = lane & 15;
    const int hi   = lane >> 4;

    // K ordering: k in [0,25) -> L taps (rows 0-2 / w1); [25,75) -> H,S taps (rows 3-5 / w2)
    // A-matrix (16x32 f16) layout: lane = M; VGPR v, half h:
    //   v<4:  K = hi*8 + v*2 + h ; v>=4: K = 16 + hi*8 + (v-4)*2 + h
    v16h A[3];
    #pragma unroll
    for (int wi = 0; wi < 3; ++wi) {
        #pragma unroll
        for (int v = 0; v < 8; ++v) {
            #pragma unroll
            for (int hb = 0; hb < 2; ++hb) {
                const int k = wi * 32 + ((v < 4) ? (hi * 8 + v * 2 + hb)
                                                 : (16 + hi * 8 + (v - 4) * 2 + hb));
                const int m = n;  // A row index
                float wv = 0.f;
                if (m < 3)       { if (k < 25)            wv = w1[m * 25 + k]; }
                else if (m < 6)  { if (k >= 25 && k < 75) wv = w2[(m - 3) * 50 + (k - 25)]; }
                A[wi][v * 2 + hb] = (_Float16)wv;
            }
        }
    }

    // B-matrix (32x16 f16) layout: lane = N (mod 16); element s: K = hi*8 + s (s<8), 16 + hi*8 + (s-8) (s>=8).
    // Precompute per-lane LDS element offsets for all 48 K-slots (relative to row base).
    int offs[48];
    #pragma unroll
    for (int wi = 0; wi < 3; ++wi) {
        #pragma unroll
        for (int s = 0; s < 16; ++s) {
            int k = wi * 32 + ((s < 8) ? (hi * 8 + s) : (16 + hi * 8 + (s - 8)));
            if (k > 74) k = 74;                 // padded K: A rows are zero there
            int c, rem;
            if (k < 25) { c = 2; rem = k; }     // L plane
            else        { const int t = k - 25; c = t / 25; rem = t % 25; } // H,S planes
            const int dy = rem / 5, dx = rem % 5;
            offs[wi * 16 + s] = (c * LR + dy) * LSTRIDE + (wx * 16 + n + dx);
        }
    }

    // Bias folded into initial accumulator: C/D lane row m = j + 8*hi
    float bias[8];
    #pragma unroll
    for (int j = 0; j < 8; ++j) {
        const int m = j + 8 * hi;
        bias[j] = (m < 3) ? b1[m] : (m < 6) ? b2[m - 3] : 0.f;
    }

    const _Float16* L = &lds[0][0][0];
    const size_t ibase = (size_t)bimg * 3 * PLANE;

    #pragma unroll 2
    for (int r = 0; r < 16; ++r) {
        const int rowbase = (wy * 16 + r) * LSTRIDE;

        v16h B0, B1, B2;
        #pragma unroll
        for (int s = 0; s < 16; ++s) B0[s] = L[rowbase + offs[s]];
        #pragma unroll
        for (int s = 0; s < 16; ++s) B1[s] = L[rowbase + offs[16 + s]];
        #pragma unroll
        for (int s = 0; s < 16; ++s) B2[s] = L[rowbase + offs[32 + s]];

        v8f acc;
        #pragma unroll
        for (int j = 0; j < 8; ++j) acc[j] = bias[j];

        acc = __builtin_amdgcn_wmma_f32_16x16x32_f16(false, A[0], false, B0, (short)0, acc, false, false);
        acc = __builtin_amdgcn_wmma_f32_16x16x32_f16(false, A[1], false, B1, (short)0, acc, false, false);
        acc = __builtin_amdgcn_wmma_f32_16x16x32_f16(false, A[2], false, B2, (short)0, acc, false, false);

        if (hi == 0) {  // lanes 0-15 hold channels 0-7 in acc[0..7]; 0-5 are valid
            const int yg = tY0 + wy * 16 + r;
            const int xg = tX0 + wx * 16 + n;
            const size_t pix = (size_t)yg * WW + xg;
            out_xl[ibase + 0 * (size_t)PLANE + pix] = acc[0];
            out_xl[ibase + 1 * (size_t)PLANE + pix] = acc[1];
            out_xl[ibase + 2 * (size_t)PLANE + pix] = acc[2];
            out_xr[ibase + 0 * (size_t)PLANE + pix] = acc[3];
            out_xr[ibase + 1 * (size_t)PLANE + pix] = acc[4];
            out_xr[ibase + 2 * (size_t)PLANE + pix] = acc[5];
        }
    }
}

extern "C" void kernel_launch(void* const* d_in, const int* in_sizes, int n_in,
                              void* d_out, int out_size, void* d_ws, size_t ws_size,
                              hipStream_t stream) {
    (void)in_sizes; (void)n_in; (void)out_size; (void)d_ws; (void)ws_size;
    const float* x  = (const float*)d_in[0];
    const float* w1 = (const float*)d_in[1];
    const float* b1 = (const float*)d_in[2];
    const float* w2 = (const float*)d_in[3];
    const float* b2 = (const float*)d_in[4];
    float* out_xl = (float*)d_out;
    float* out_xr = out_xl + (size_t)32 * 3 * PLANE;

    dim3 grid(WW / TW, HH / TH, 32);   // (8, 16, 32)
    hsl_dualconv_wmma<<<grid, 256, 0, stream>>>(x, w1, b1, w2, b2, out_xl, out_xr);
}
